// GPTBigCodeAttention_64862596104618
// MI455X (gfx1250) — matrix operand order repaired and checked by reading, not compiled, not run on hardware
//
#include <hip/hip_runtime.h>

#define BB 2
#define SS 2048
#define EE 2048
#define HH 16
#define DD 128

typedef __attribute__((ext_vector_type(16))) __bf16 v16bf;
typedef __attribute__((ext_vector_type(8)))  float  v8f;

typedef __attribute__((ext_vector_type(4))) unsigned int tdm_u32x4;
typedef __attribute__((ext_vector_type(8))) int          tdm_i32x8;
typedef __attribute__((ext_vector_type(4))) int          tdm_i32x4;

__device__ __forceinline__ unsigned short f32_to_bf16(float f) {
  unsigned int u = __float_as_uint(f);
  u += 0x7FFFu + ((u >> 16) & 1u);   // round-to-nearest-even
  return (unsigned short)(u >> 16);
}

// Build a 16-element bf16 fragment from two 16-byte chunks.
__device__ __forceinline__ v16bf frag_ld(const unsigned short* p0, const unsigned short* p1) {
  union { v16bf v; unsigned int w[8]; } u;
  uint4 a = *reinterpret_cast<const uint4*>(p0);
  uint4 b = *reinterpret_cast<const uint4*>(p1);
  u.w[0] = a.x; u.w[1] = a.y; u.w[2] = a.z; u.w[3] = a.w;
  u.w[4] = b.x; u.w[5] = b.y; u.w[6] = b.z; u.w[7] = b.w;
  return u.v;
}

// Tensor Data Mover: DMA a 2D row-major tile of 2-byte elements into LDS.
// D# bitfields per CDNA5 ISA 8.3/8.4 (group0: control/lds/global addr, type=2;
// group1: data_size=2B, tensor dims, tile dims, dim0 stride).
__device__ __forceinline__ void tdm_load_2d_b16(unsigned lds_off, const void* gptr,
                                                unsigned tile_w, unsigned tile_h,
                                                unsigned row_stride_elems) {
  unsigned long long ga = (unsigned long long)gptr;
  tdm_u32x4 g0;
  g0[0] = 1u;                                              // count=1, user descriptor
  g0[1] = lds_off;                                         // lds_addr (bytes)
  g0[2] = (unsigned)(ga & 0xFFFFFFFFu);                    // global_addr[31:0]
  g0[3] = (unsigned)((ga >> 32) & 0x1FFFFFFu) | (2u << 30);// global_addr[56:32], type=2
  tdm_i32x8 g1;
  g1[0] = (int)(1u << 16);                                 // wg_mask=0, data_size=1 (2B)
  g1[1] = (int)(tile_w << 16);                             // tensor_dim0[15:0] @ bits63:48
  g1[2] = (int)((tile_w >> 16) | (tile_h << 16));          // tdim0 hi | tensor_dim1 lo
  g1[3] = (int)((tile_h >> 16) | (tile_w << 16));          // tdim1 hi | tile_dim0
  g1[4] = (int)tile_h;                                     // tile_dim1 (tile_dim2=0)
  g1[5] = (int)row_stride_elems;                           // tensor_dim0_stride[31:0]
  g1[6] = 0; g1[7] = 0;                                    // stride hi, dim1_stride=0
  tdm_i32x4 z4 = {0, 0, 0, 0};
#if __clang_major__ >= 23
  tdm_i32x8 z8 = {0, 0, 0, 0, 0, 0, 0, 0};
  __builtin_amdgcn_tensor_load_to_lds(g0, g1, z4, z4, z8, 0);
#else
  __builtin_amdgcn_tensor_load_to_lds(g0, g1, z4, z4, 0);
#endif
}

__global__ void cvt_f32_bf16(const float* __restrict__ s, unsigned short* __restrict__ d, int n) {
  int i = blockIdx.x * blockDim.x + threadIdx.x;
  if (i < n) d[i] = f32_to_bf16(s[i]);
}

__global__ void qkv_split(const float* __restrict__ qkv, unsigned short* __restrict__ qb,
                          unsigned short* __restrict__ kb, unsigned short* __restrict__ vb, int n) {
  int i = blockIdx.x * blockDim.x + threadIdx.x;
  if (i >= n) return;
  const int NQ = EE + 2 * DD;
  int row = i / NQ;
  int col = i - row * NQ;
  unsigned short v = f32_to_bf16(qkv[i]);
  if (col < EE)           qb[(size_t)row * EE + col] = v;
  else if (col < EE + DD) kb[(size_t)row * DD + (col - EE)] = v;
  else                    vb[(size_t)row * DD + (col - EE - DD)] = v;
}

// C[M,N] = A[M,K](bf16) * B[K,N](bf16) + bias[N], fp32 out.
// Block: 256 threads (8 waves), tile 128x128, wave sub-tile 32x64 (2x4 WMMA).
__global__ __launch_bounds__(256) void gemm_bf16_wmma(
    const unsigned short* __restrict__ A, const unsigned short* __restrict__ B,
    const float* __restrict__ bias, float* __restrict__ C, int M, int N, int K)
{
  __shared__ __align__(16) unsigned short As[128][32];    // 8 KB
  __shared__ __align__(16) unsigned short Bst[128][32];   // 8 KB, transposed [n][k]
  const int t = threadIdx.x;
  const int lane = t & 31, wave = t >> 5;
  const int wm = wave & 3, wn = wave >> 2;                // 4x2 wave grid
  const int lm = lane & 15, lh = lane >> 4;
  const int mbase = blockIdx.y * 128, nbase = blockIdx.x * 128;

  v8f acc[2][4] = {};

  const int arow = t >> 1, acol = (t & 1) * 16;
  const int brow = t >> 3, bcol = (t & 7) * 16;

  for (int kt = 0; kt < K; kt += 32) {
    const uint4* pa = reinterpret_cast<const uint4*>(&A[(size_t)(mbase + arow) * K + kt + acol]);
    uint4 a0 = pa[0], a1 = pa[1];
    const uint4* pb = reinterpret_cast<const uint4*>(&B[(size_t)(kt + brow) * N + nbase + bcol]);
    uint4 b0 = pb[0], b1 = pb[1];
    if (kt + 32 < K) {
      __builtin_prefetch(&A[(size_t)(mbase + arow) * K + kt + 32 + acol], 0, 1);
      __builtin_prefetch(&B[(size_t)(kt + 32 + brow) * N + nbase + bcol], 0, 1);
    }
    *reinterpret_cast<uint4*>(&As[arow][acol])     = a0;
    *reinterpret_cast<uint4*>(&As[arow][acol + 8]) = a1;
    union { unsigned int w[8]; unsigned short s8[16]; } ub;
    ub.w[0] = b0.x; ub.w[1] = b0.y; ub.w[2] = b0.z; ub.w[3] = b0.w;
    ub.w[4] = b1.x; ub.w[5] = b1.y; ub.w[6] = b1.z; ub.w[7] = b1.w;
    #pragma unroll
    for (int e = 0; e < 16; ++e) Bst[bcol + e][brow] = ub.s8[e];
    __syncthreads();

    v16bf af[2], bf[4];
    #pragma unroll
    for (int i = 0; i < 2; ++i)
      af[i] = frag_ld(&As[wm*32 + i*16 + lm][lh*8], &As[wm*32 + i*16 + lm][16 + lh*8]);
    #pragma unroll
    for (int j = 0; j < 4; ++j)
      bf[j] = frag_ld(&Bst[wn*64 + j*16 + lm][lh*8], &Bst[wn*64 + j*16 + lm][16 + lh*8]);
    #pragma unroll
    for (int i = 0; i < 2; ++i)
      #pragma unroll
      for (int j = 0; j < 4; ++j)
        acc[i][j] = __builtin_amdgcn_wmma_f32_16x16x32_bf16(
            false, af[i], false, bf[j], (short)0, acc[i][j], false, false);
    __syncthreads();
  }

  #pragma unroll
  for (int i = 0; i < 2; ++i)
    #pragma unroll
    for (int j = 0; j < 4; ++j) {
      const int col = nbase + wn*64 + j*16 + lm;
      const float bc = bias[col];
      #pragma unroll
      for (int v = 0; v < 8; ++v) {
        const int row = mbase + wm*32 + i*16 + v + 8*lh;
        C[(size_t)row * N + col] = acc[i][j][v] + bc;
      }
    }
}

// Fused multi-query causal flash attention.
// Grid: (S/64, H, B). Block: 128 threads (4 waves), wave owns 16 q-rows.
// K chunk DMA'd into LDS by the Tensor Data Mover; V transposed cooperatively.
__global__ __launch_bounds__(128) void mqa_flash(
    const unsigned short* __restrict__ qb, const unsigned short* __restrict__ kb,
    const unsigned short* __restrict__ vb, unsigned short* __restrict__ ob)
{
  __shared__ __align__(16) unsigned short Ks[32][DD];    // 8 KB, [key][d]
  __shared__ __align__(16) unsigned short Vst[DD][32];   // 8 KB, transposed [d][key]
  __shared__ float Sw[4][16][32];                        // 8 KB per-wave score scratch
  __shared__ __align__(16) unsigned short Pw[4][16][32]; // 4 KB probabilities (bf16)
  __shared__ float alphaS[4][16];
  __shared__ float rinvS[4][16];

  const int t = threadIdx.x;
  const int lane = t & 31, wv = t >> 5;
  const int lm = lane & 15, lh = lane >> 4;
  const int qbase = blockIdx.x * 64;
  const int head = blockIdx.y;
  const int b = blockIdx.z;
  const int qi = qbase + wv*16 + lm;

  // Q tile (16x128) held in registers for whole kernel: 4 A-fragments.
  const unsigned short* qrow = qb + (size_t)(b*SS + qi) * EE + head * DD;
  v16bf qf[4];
  #pragma unroll
  for (int s = 0; s < 4; ++s)
    qf[s] = frag_ld(qrow + s*32 + lh*8, qrow + s*32 + 16 + lh*8);

  const unsigned ks_lds_off = (unsigned)(unsigned long long)(const void*)&Ks[0][0];

  v8f accO[8] = {};
  float mrun = -1e30f, lrun = 0.0f;
  const float sc = 0.08838834764831845f;  // (1/(layer+1)/sqrt(D)) * (layer+1) = 1/sqrt(128)

  const int nch = (qbase + 64) >> 5;      // causal: keys up to qbase+63
  for (int ch = 0; ch < nch; ++ch) {
    const int kbase = ch * 32;
    __syncthreads();
    if (t == 0)  // TDM: DMA 32x128 bf16 K tile into LDS (EXEC ignored by TDM)
      tdm_load_2d_b16(ks_lds_off, kb + (size_t)(b*SS + kbase) * DD, DD, 32, DD);
    { // cooperative V chunk load, transposed (128 threads)
      const int rt = t >> 2, cseg = (t & 3) * 32;
      const uint4* vp = reinterpret_cast<const uint4*>(vb + (size_t)(b*SS + kbase + rt) * DD + cseg);
      #pragma unroll
      for (int i2 = 0; i2 < 4; ++i2) {
        uint4 vv = vp[i2];
        union { unsigned int w[4]; unsigned short s8[8]; } u;
        u.w[0] = vv.x; u.w[1] = vv.y; u.w[2] = vv.z; u.w[3] = vv.w;
        #pragma unroll
        for (int e = 0; e < 8; ++e) Vst[cseg + i2*8 + e][rt] = u.s8[e];
      }
    }
    __builtin_amdgcn_s_wait_tensorcnt(0);
    __syncthreads();

    // S(16x32) = Q(16x128) * K^T : 8 WMMAs over D
    v8f accS[2] = {};
    #pragma unroll
    for (int s = 0; s < 4; ++s)
      #pragma unroll
      for (int j = 0; j < 2; ++j) {
        v16bf kf = frag_ld(&Ks[j*16 + lm][s*32 + lh*8], &Ks[j*16 + lm][s*32 + 16 + lh*8]);
        accS[j] = __builtin_amdgcn_wmma_f32_16x16x32_bf16(
            false, qf[s], false, kf, (short)0, accS[j], false, false);
      }
    #pragma unroll
    for (int j = 0; j < 2; ++j)
      #pragma unroll
      for (int v = 0; v < 8; ++v)
        Sw[wv][v + 8*lh][j*16 + lm] = accS[j][v];
    __syncthreads();

    // Online softmax: lanes 0-15 of each wave own one row each.
    if (lh == 0) {
      const int r = lm;
      const int qr = qbase + wv*16 + r;
      float rmax = -1e30f;
      #pragma unroll
      for (int c = 0; c < 32; ++c) {
        float sv = Sw[wv][r][c] * sc;
        if (kbase + c > qr) sv = -1e30f;   // causal mask
        Sw[wv][r][c] = sv;
        rmax = fmaxf(rmax, sv);
      }
      const float mnew  = fmaxf(mrun, rmax);
      const float alpha = __expf(mrun - mnew);
      float psum = 0.0f;
      #pragma unroll
      for (int c = 0; c < 32; ++c) {
        float p = __expf(Sw[wv][r][c] - mnew);
        psum += p;
        Pw[wv][r][c] = f32_to_bf16(p);
      }
      lrun = alpha * lrun + psum;
      mrun = mnew;
      alphaS[wv][r] = alpha;
    }
    __syncthreads();

    // O = alpha*O + P(16x32) * V(32x128): 8 WMMAs over D columns
    v16bf pf = frag_ld(&Pw[wv][lm][lh*8], &Pw[wv][lm][16 + lh*8]);
    float fac[8];
    #pragma unroll
    for (int v = 0; v < 8; ++v) fac[v] = alphaS[wv][v + 8*lh];
    #pragma unroll
    for (int d = 0; d < 8; ++d) {
      #pragma unroll
      for (int v = 0; v < 8; ++v) accO[d][v] *= fac[v];
      v16bf vf = frag_ld(&Vst[d*16 + lm][lh*8], &Vst[d*16 + lm][16 + lh*8]);
      accO[d] = __builtin_amdgcn_wmma_f32_16x16x32_bf16(
          false, pf, false, vf, (short)0, accO[d], false, false);
    }
  }

  if (lh == 0) rinvS[wv][lm] = 1.0f / lrun;
  __syncthreads();
  float rv[8];
  #pragma unroll
  for (int v = 0; v < 8; ++v) rv[v] = rinvS[wv][v + 8*lh];
  #pragma unroll
  for (int d = 0; d < 8; ++d) {
    const int col = head*DD + d*16 + lm;
    #pragma unroll
    for (int v = 0; v < 8; ++v) {
      const int row = qbase + wv*16 + v + 8*lh;
      ob[(size_t)(b*SS + row) * EE + col] = f32_to_bf16(accO[d][v] * rv[v]);
    }
  }
}

extern "C" void kernel_launch(void* const* d_in, const int* in_sizes, int n_in,
                              void* d_out, int out_size, void* d_ws, size_t ws_size,
                              hipStream_t stream) {
  (void)in_sizes; (void)n_in; (void)out_size; (void)ws_size;
  const float* hid   = (const float*)d_in[0];
  const float* wattn = (const float*)d_in[1];
  const float* battn = (const float*)d_in[2];
  const float* wproj = (const float*)d_in[3];
  const float* bproj = (const float*)d_in[4];
  // d_in[5] attention_mask: causal tril, applied analytically in-kernel.
  float* out = (float*)d_out;

  const size_t rows = (size_t)BB * SS;     // 4096
  const size_t NQ   = EE + 2 * DD;         // 2304

  char* ws = (char*)d_ws;
  size_t off = 0;
  auto alloc = [&](size_t bytes) { size_t p = off; off = (off + bytes + 255) & ~(size_t)255; return p; };
  unsigned short* hid_bf = (unsigned short*)(ws + alloc(rows * EE * 2));
  unsigned short* wat_bf = (unsigned short*)(ws + alloc((size_t)EE * NQ * 2));
  unsigned short* wpr_bf = (unsigned short*)(ws + alloc((size_t)EE * EE * 2));
  float*          qkv    = (float*)         (ws + alloc(rows * NQ * 4));
  unsigned short* q_bf   = (unsigned short*)(ws + alloc(rows * EE * 2));
  unsigned short* k_bf   = (unsigned short*)(ws + alloc(rows * DD * 2));
  unsigned short* v_bf   = (unsigned short*)(ws + alloc(rows * DD * 2));
  unsigned short* at_bf  = (unsigned short*)(ws + alloc(rows * EE * 2));

  int n1 = (int)(rows * EE);
  cvt_f32_bf16<<<(n1 + 255) / 256, 256, 0, stream>>>(hid, hid_bf, n1);
  int n2 = (int)((size_t)EE * NQ);
  cvt_f32_bf16<<<(n2 + 255) / 256, 256, 0, stream>>>(wattn, wat_bf, n2);
  int n3 = EE * EE;
  cvt_f32_bf16<<<(n3 + 255) / 256, 256, 0, stream>>>(wproj, wpr_bf, n3);

  gemm_bf16_wmma<<<dim3((unsigned)(NQ / 128), (unsigned)(rows / 128)), 256, 0, stream>>>(
      hid_bf, wat_bf, battn, qkv, (int)rows, (int)NQ, EE);

  int n4 = (int)(rows * NQ);
  qkv_split<<<(n4 + 255) / 256, 256, 0, stream>>>(qkv, q_bf, k_bf, v_bf, n4);

  mqa_flash<<<dim3(SS / 64, HH, BB), 128, 0, stream>>>(q_bf, k_bf, v_bf, at_bf);

  gemm_bf16_wmma<<<dim3(EE / 128, (unsigned)(rows / 128)), 256, 0, stream>>>(
      at_bf, wpr_bf, bproj, out, (int)rows, EE, EE);
}
